// ResidualSNNBlock_4776003633777
// MI455X (gfx1250) — matrix operand order repaired and checked by reading, not compiled, not run on hardware
//
#include <hip/hip_runtime.h>

// ---------------------------------------------------------------------------
// Types for WMMA bf16 (probe-verified builtin signature):
//   __builtin_amdgcn_wmma_f32_16x16x32_bf16(bool, v16bf, bool, v16bf,
//                                           short, v8f, bool, bool)
// ---------------------------------------------------------------------------
typedef __attribute__((ext_vector_type(16))) __bf16 v16bf;
typedef __attribute__((ext_vector_type(8)))  __bf16 v8bf;
typedef __attribute__((ext_vector_type(8)))  float  v8f;

#define TT 8
#define BB 16
#define NN (TT*BB)          // 128 merged batch
#define CC 64
#define HH 64
#define WW 64
#define HWC (HH*WW*CC)      // 262144 elements per image (NHWC)
#define CHW (CC*HH*WW)      // 262144 elements per image (NCHW)
#define HW  (HH*WW)         // 4096
#define BN_N ((float)(NN*HH*WW))   // 524288 reduction count per channel

union FragAB { v16bf v; uint4 q[2]; };
union Pack8  { v8bf  v; uint4 q;    };

// ---------------------------------------------------------------------------
// Kernel 1: pack conv weights OIHW f32 -> [tap][co][ci] bf16 ; zero BN stats
// ---------------------------------------------------------------------------
__global__ void pack_w_kernel(const float* __restrict__ cw,
                              __bf16* __restrict__ wt,
                              float* __restrict__ stats) {
    int g = blockIdx.x * 256 + threadIdx.x;
    if (g < 9 * CC * CC) {
        int pos = g / (CC * CC);       // tap index r*3+s
        int co  = (g >> 6) & 63;
        int ci  = g & 63;
        wt[g] = (__bf16)cw[(co * CC + ci) * 9 + pos];
    }
    if (g < 256) stats[g] = 0.0f;      // sum[64], sumsq[64], inv[64], bias[64]
}

// ---------------------------------------------------------------------------
// Kernel 2: NCHW f32 -> NHWC bf16 repack (LDS-tiled (c,w) transpose per row)
// ---------------------------------------------------------------------------
__global__ void pack_x_kernel(const float* __restrict__ x,
                              __bf16* __restrict__ xn) {
    __shared__ __bf16 tile[CC * 65];
    int blk = blockIdx.x;              // n*HH + h
    int n = blk >> 6, h = blk & 63;
    int tid = threadIdx.x;

    #pragma unroll
    for (int i = 0; i < 16; ++i) {     // coalesced read along w
        int idx = tid + i * 256;       // c*64 + w
        int c = idx >> 6, w = idx & 63;
        tile[c * 65 + w] = (__bf16)x[(size_t)n * CHW + (size_t)c * HW + h * WW + w];
    }
    __syncthreads();
    #pragma unroll
    for (int i = 0; i < 16; ++i) {     // coalesced write along c
        int idx = tid + i * 256;       // w*64 + c
        int w = idx >> 6, c = idx & 63;
        xn[(size_t)n * HWC + (size_t)h * (WW * CC) + w * CC + c] = tile[c * 65 + w];
    }
}

// ---------------------------------------------------------------------------
// Kernel 3: implicit-GEMM 3x3 conv via v_wmma_f32_16x16x32_bf16.
// Block = 128 threads (4 waves). Tile: one (n,h) row -> M=64 pixels, N=64 co.
// Wave wv owns co in [wv*16, wv*16+16), accumulates 4 M-tiles.
// Fused epilogue: per-channel sum / sumsq atomics + bf16 NCHW store of y.
// ---------------------------------------------------------------------------
__global__ void conv_wmma_kernel(const __bf16* __restrict__ xn,
                                 const __bf16* __restrict__ wt,
                                 __bf16* __restrict__ ybf,
                                 float* __restrict__ stats) {
    __shared__ __bf16 lds[3 * 66 * CC];          // 3 padded rows, 25344 B
    const int tid = threadIdx.x;
    const int blk = blockIdx.x;                  // n*HH + h
    const int n = blk >> 6, hrow = blk & 63;

    // ---- stage 3 input rows (with zero halo) into LDS -----------------
    #pragma unroll
    for (int r = 0; r < 3; ++r) {
        int hr = hrow + r - 1;
        uint4* dst = (uint4*)&lds[(r * 66 + 1) * CC];
        if (hr >= 0 && hr < HH) {
            const uint4* src = (const uint4*)(xn + (size_t)n * HWC + (size_t)hr * (WW * CC));
            #pragma unroll
            for (int j = 0; j < 4; ++j) dst[tid + j * 128] = src[tid + j * 128];
        } else {
            uint4 z = {0u, 0u, 0u, 0u};
            #pragma unroll
            for (int j = 0; j < 4; ++j) dst[tid + j * 128] = z;
        }
    }
    if (tid < 48) {                              // zero halo columns p=0, p=65
        int r = tid >> 4, k = tid & 15;
        int p = (k & 8) ? 65 : 0;
        uint4 z = {0u, 0u, 0u, 0u};
        *(uint4*)&lds[(r * 66 + p) * CC + (k & 7) * 8] = z;
    }
    __syncthreads();

    const int wv   = tid >> 5;
    const int lane = tid & 31;
    const int co0  = wv * 16;
    const int nIdx = lane & 15;                  // N index of this lane
    const int kg   = lane >> 4;                  // K half-group
    const int co   = co0 + nIdx;

    v8f acc[4] = {};                             // 4 M-tiles of fp32 C/D

    #pragma unroll
    for (int r = 0; r < 3; ++r) {
        #pragma unroll
        for (int s = 0; s < 3; ++s) {
            const int pos = r * 3 + s;
            #pragma unroll
            for (int kc = 0; kc < 2; ++kc) {     // ci chunks of 32
                // B fragment: 16 contiguous ci at fixed co (32 B load)
                FragAB bfr;
                const uint4* wp = (const uint4*)(wt + ((size_t)(pos * CC + co)) * CC
                                                    + kc * 32 + kg * 16);
                bfr.q[0] = wp[0];
                bfr.q[1] = wp[1];
                #pragma unroll
                for (int mt = 0; mt < 4; ++mt) {
                    // A fragment from LDS: 16 contiguous ci per K half-group
                    FragAB afr;
                    const uint4* ap = (const uint4*)&lds[(r * 66 + mt * 16 + nIdx + s) * CC
                                                         + kc * 32 + kg * 16];
                    afr.q[0] = ap[0];
                    afr.q[1] = ap[1];
                    acc[mt] = __builtin_amdgcn_wmma_f32_16x16x32_bf16(
                        false, afr.v, false, bfr.v, (short)0, acc[mt], false, false);
                }
            }
        }
    }

    // ---- fused BN statistics: per-channel sum / sumsq -----------------
    float ssum = 0.0f, ssq = 0.0f;
    #pragma unroll
    for (int mt = 0; mt < 4; ++mt)
        #pragma unroll
        for (int e = 0; e < 8; ++e) {
            float f = acc[mt][e];
            ssum += f; ssq += f * f;
        }
    ssum += __shfl_xor(ssum, 16, 32);            // combine lane L with L+16 (same co)
    ssq  += __shfl_xor(ssq,  16, 32);
    if (lane < 16) {
        atomicAdd(&stats[co], ssum);             // v_global_atomic_add_f32
        atomicAdd(&stats[64 + co], ssq);
    }

    // ---- store y as bf16 NCHW -----------------------------------------
    #pragma unroll
    for (int mt = 0; mt < 4; ++mt) {
        Pack8 pk;
        #pragma unroll
        for (int e = 0; e < 8; ++e) pk.v[e] = (__bf16)acc[mt][e];
        int w0 = mt * 16 + 8 * kg;               // M = e + 8*kg within tile
        *(uint4*)(ybf + (size_t)n * CHW + (size_t)co * HW + hrow * WW + w0) = pk.q;
    }
}

// ---------------------------------------------------------------------------
// Kernel 4: finalize BN -> per-channel scale (inv) and shift (bias)
// ---------------------------------------------------------------------------
__global__ void bn_finalize_kernel(float* __restrict__ stats,
                                   const float* __restrict__ gamma,
                                   const float* __restrict__ beta) {
    int c = threadIdx.x;
    if (c < CC) {
        const float Ninv = 1.0f / BN_N;
        float mean = stats[c] * Ninv;
        float var  = stats[64 + c] * Ninv - mean * mean;
        float inv  = gamma[c] * rsqrtf(var + 1e-5f);
        stats[128 + c] = inv;
        stats[192 + c] = beta[c] - mean * inv;
    }
}

// ---------------------------------------------------------------------------
// Kernel 5: multi-step LIF scan over T + residual. One thread per (b,c,h,w).
// v' = (v + y)/2 (tau=2); spike = (v' >= 1); hard reset; out = x + spike.
// ---------------------------------------------------------------------------
__global__ void lif_kernel(const float* __restrict__ x,
                           const __bf16* __restrict__ ybf,
                           const float* __restrict__ stats,
                           float* __restrict__ out) {
    int g = blockIdx.x * 256 + threadIdx.x;      // 0 .. 4194303
    int hw = g & (HW - 1);
    int c  = (g >> 12) & 63;
    int b  = g >> 18;
    float inv  = stats[128 + c];
    float bias = stats[192 + c];
    float v = 0.0f;
    size_t base = (size_t)b * CHW + (size_t)c * HW + hw;
    const size_t tstride = (size_t)BB * CHW;     // +16 images per timestep
    #pragma unroll
    for (int t = 0; t < TT; ++t) {
        size_t idx = base + (size_t)t * tstride;
        if (t < TT - 1)
            __builtin_prefetch(ybf + idx + tstride, 0, 1);   // global_prefetch_b8
        float y = (float)ybf[idx] * inv + bias;
        float h = 0.5f * (v + y);
        float s = (h >= 1.0f) ? 1.0f : 0.0f;
        v = (1.0f - s) * h;
        out[idx] = x[idx] + s;
    }
}

// ---------------------------------------------------------------------------
// Launch: ws layout: [wt bf16 73728 B][stats f32 1024 B][xn bf16 64 MiB][y bf16 64 MiB]
// ---------------------------------------------------------------------------
extern "C" void kernel_launch(void* const* d_in, const int* in_sizes, int n_in,
                              void* d_out, int out_size, void* d_ws, size_t ws_size,
                              hipStream_t stream) {
    (void)in_sizes; (void)n_in; (void)out_size; (void)ws_size;
    const float* x     = (const float*)d_in[0];
    const float* cw    = (const float*)d_in[1];
    const float* gamma = (const float*)d_in[2];
    const float* beta  = (const float*)d_in[3];
    float* out = (float*)d_out;

    char* ws = (char*)d_ws;
    __bf16* wt    = (__bf16*)(ws);
    float*  stats = (float*)(ws + 73728);
    __bf16* xn    = (__bf16*)(ws + 74752);
    __bf16* ybf   = (__bf16*)(ws + 74752 + (size_t)NN * HWC * 2);

    pack_w_kernel     <<<144,   256, 0, stream>>>(cw, wt, stats);
    pack_x_kernel     <<<NN*HH, 256, 0, stream>>>(x, xn);
    conv_wmma_kernel  <<<NN*HH, 128, 0, stream>>>(xn, wt, ybf, stats);
    bn_finalize_kernel<<<1,      64, 0, stream>>>(stats, gamma, beta);
    lif_kernel        <<<16384, 256, 0, stream>>>(x, ybf, stats, out);
}
